// SelfAttention_13821204758999
// MI455X (gfx1250) — matrix operand order repaired and checked by reading, not compiled, not run on hardware
//
#include <hip/hip_runtime.h>

typedef __attribute__((ext_vector_type(16))) _Float16 v16h;
typedef __attribute__((ext_vector_type(8)))  _Float16 v8h;
typedef __attribute__((ext_vector_type(8)))  float    v8f;

#define GAMMA  0.5f
#define C_DIM  64
#define N_DIM  4096
#define B_DIM  4
#define NT     32          // keys per iteration
#define WPB    2           // waves per block, each wave owns 32 queries
#define PT_STR 40          // P bounce-tile row stride (halves, multiple of 8)

static __device__ inline v16h cat8(v8h a, v8h b) {
    return __builtin_shufflevector(a, b, 0,1,2,3,4,5,6,7,8,9,10,11,12,13,14,15);
}
static __device__ inline v8f wmma16(v16h a, v16h b, v8f c) {
    return __builtin_amdgcn_wmma_f32_16x16x32_f16(false, a, false, b, (short)0, c, false, false);
}

// ---------------- pre-pass: x (f32) -> xh (f16, [b][c][n]) + xt (f16, [b][n][c]) ----
__global__ __launch_bounds__(256)
void cvt_kernel(const float* __restrict__ x,
                _Float16* __restrict__ xh, _Float16* __restrict__ xt)
{
    __shared__ __align__(16) _Float16 ts[64 * 72];   // 64n x 64c tile, padded
    const int tid = threadIdx.x;
    const int b   = blockIdx.x >> 6;
    const int n0  = (blockIdx.x & 63) * 64;
    {
        const int c  = tid >> 2;
        const int jb = (tid & 3) * 16;
        const float* src = x + ((size_t)b * C_DIM + c) * N_DIM + n0 + jb;
        _Float16 h[16];
        #pragma unroll
        for (int i = 0; i < 16; ++i) h[i] = (_Float16)src[i];
        v8h v0, v1;
        #pragma unroll
        for (int i = 0; i < 8; ++i) { v0[i] = h[i]; v1[i] = h[8 + i]; }
        _Float16* dst = xh + ((size_t)b * C_DIM + c) * N_DIM + n0 + jb;
        *(v8h*)dst = v0; *(v8h*)(dst + 8) = v1;
        #pragma unroll
        for (int i = 0; i < 16; ++i) ts[(jb + i) * 72 + c] = h[i];
    }
    __syncthreads();
    {
        const int n  = tid >> 2;
        const int cb = (tid & 3) * 16;
        v8h v0 = *(v8h*)&ts[n * 72 + cb];
        v8h v1 = *(v8h*)&ts[n * 72 + cb + 8];
        _Float16* dst = xt + ((size_t)b * N_DIM + n0 + n) * C_DIM + cb;
        *(v8h*)dst = v0; *(v8h*)(dst + 8) = v1;
    }
}

// ---------------- flash attention, waves independent, no hot-loop barriers ---------
__global__ __launch_bounds__(32 * WPB)
void fa_kernel(const _Float16* __restrict__ xh, const _Float16* __restrict__ xt,
               const float* __restrict__ x, float* __restrict__ out)
{
    __shared__ __align__(16) _Float16 pt[WPB * 2 * 16 * PT_STR];  // per-wave P bounce

    const int lane = threadIdx.x & 31;
    const int wave = threadIdx.x >> 5;
    const int half = lane >> 4;
    const int lrow = lane & 15;

    const int bid = blockIdx.x;
    const int b   = bid >> 6;                 // 64 blocks per batch
    const int m0  = (bid & 63) * (WPB * 32);  // 64 queries per block

    const _Float16* xhb = xh + (size_t)b * C_DIM * N_DIM;   // [c][n]
    const _Float16* xtb = xt + (size_t)b * N_DIM * C_DIM;   // [n][c]

    // ---- Q B-fragments, straight from transposed f16 (kept all kernel) ----
    // B[k=c, j=m]: lane j=lrow, K = 8*half + 0..7, 16 + 8*half + 0..7 (+32 for frag 1)
    const int mbase = m0 + wave * 32;
    v16h bq[2][2];
    #pragma unroll
    for (int mt = 0; mt < 2; ++mt) {
        const _Float16* qrow = xtb + (size_t)(mbase + mt * 16 + lrow) * C_DIM;
        #pragma unroll
        for (int f = 0; f < 2; ++f) {
            const _Float16* p = qrow + f * 32 + 8 * half;
            bq[mt][f] = cat8(*(const v8h*)p, *(const v8h*)(p + 16));
        }
    }

    v8f   oacc[2][4] = {};                    // [mt][ct]: O[c,m] D-tiles
    float rmax[2] = {-3.0e38f, -3.0e38f};
    float rsum[2] = {0.0f, 0.0f};
    _Float16* myPt = &pt[wave * 2 * 16 * PT_STR];

    for (int n0 = 0; n0 < N_DIM; n0 += NT) {
        // prefetch next key/value tile lines into cache
        if (n0 + NT < N_DIM) {
            __builtin_prefetch(xtb + (size_t)(n0 + NT + lane) * C_DIM, 0, 0);
            __builtin_prefetch(xhb + (size_t)(2 * lane)     * N_DIM + n0 + NT, 0, 0);
            __builtin_prefetch(xhb + (size_t)(2 * lane + 1) * N_DIM + n0 + NT, 0, 0);
        }

        // ---- scores S[n,m]: A = K^T rows n (from xt), B = Q ----
        v8f s[2][2];
        #pragma unroll
        for (int t = 0; t < 2; ++t) {
            const _Float16* krow = xtb + (size_t)(n0 + t * 16 + lrow) * C_DIM;
            v16h a0 = cat8(*(const v8h*)(krow + 8 * half),
                           *(const v8h*)(krow + 8 * half + 16));       // c 0..31
            v16h a1 = cat8(*(const v8h*)(krow + 32 + 8 * half),
                           *(const v8h*)(krow + 32 + 8 * half + 16));  // c 32..63
            #pragma unroll
            for (int mt = 0; mt < 2; ++mt) {
                v8f acc = {};
                acc = wmma16(a0, bq[mt][0], acc);
                acc = wmma16(a1, bq[mt][1], acc);
                s[t][mt] = acc;
            }
        }

        // ---- online softmax over n (column-wise in D layout), per m-tile ----
        #pragma unroll
        for (int mt = 0; mt < 2; ++mt) {
            float tmax = s[0][mt][0];
            #pragma unroll
            for (int v = 1; v < 8; ++v) tmax = fmaxf(tmax, s[0][mt][v]);
            #pragma unroll
            for (int v = 0; v < 8; ++v) tmax = fmaxf(tmax, s[1][mt][v]);
            tmax = fmaxf(tmax, __shfl_xor(tmax, 16, 32));
            const float nmax = fmaxf(rmax[mt], tmax);
            const float corr = __expf(rmax[mt] - nmax);
            rmax[mt] = nmax;

            float tsum = 0.0f;
            #pragma unroll
            for (int t = 0; t < 2; ++t) {
                v8h ph;
                #pragma unroll
                for (int v = 0; v < 8; ++v) {
                    const float p = __expf(s[t][mt][v] - nmax);
                    tsum += p;
                    ph[v] = (_Float16)p;
                }
                // lane holds (m=lrow, n = 16t + 8*half + v): contiguous chunk
                *(v8h*)&myPt[(mt * 16 + lrow) * PT_STR + t * 16 + 8 * half] = ph;
            }
            tsum += __shfl_xor(tsum, 16, 32);
            rsum[mt] = rsum[mt] * corr + tsum;
            #pragma unroll
            for (int ct = 0; ct < 4; ++ct)
                #pragma unroll
                for (int v = 0; v < 8; ++v) oacc[mt][ct][v] *= corr;
        }

        // per-wave cross-lane LDS RAW: drain DS before reading P back
        asm volatile("s_wait_dscnt 0" ::: "memory");

        // ---- P B-fragments (B[k=n, j=m]; lane: j=lrow, K = 8*half + 0..7, 16 + 8*half + 0..7) ----
        v16h bp[2];
        #pragma unroll
        for (int mt = 0; mt < 2; ++mt) {
            const _Float16* pp = &myPt[(mt * 16 + lrow) * PT_STR + 8 * half];
            bp[mt] = cat8(*(const v8h*)pp, *(const v8h*)(pp + 16));
        }

        // ---- O[c,m] += V (A rows c, from xh) x P; V frag shared by both m-tiles ----
        #pragma unroll
        for (int ct = 0; ct < 4; ++ct) {
            const _Float16* vrow = xhb + (size_t)(ct * 16 + lrow) * N_DIM + n0;
            v16h av = cat8(*(const v8h*)(vrow + 8 * half),
                           *(const v8h*)(vrow + 8 * half + 16));
            #pragma unroll
            for (int mt = 0; mt < 2; ++mt)
                oacc[mt][ct] = wmma16(av, bp[mt], oacc[mt][ct]);
        }
    }

    // ---- epilogue: out = gamma * O / rsum + x ----
    #pragma unroll
    for (int mt = 0; mt < 2; ++mt) {
        const float inv = GAMMA / rsum[mt];
        const int   mg  = mbase + mt * 16 + lrow;
        #pragma unroll
        for (int ct = 0; ct < 4; ++ct) {
            #pragma unroll
            for (int v = 0; v < 8; ++v) {
                const int c = ct * 16 + v + 8 * half;
                const size_t idx = ((size_t)b * C_DIM + c) * N_DIM + mg;
                out[idx] = oacc[mt][ct][v] * inv + x[idx];
            }
        }
    }
}

extern "C" void kernel_launch(void* const* d_in, const int* in_sizes, int n_in,
                              void* d_out, int out_size, void* d_ws, size_t ws_size,
                              hipStream_t stream) {
    (void)in_sizes; (void)n_in; (void)out_size; (void)ws_size;  // needs ws >= 4 MB
    const float* x   = (const float*)d_in[0];
    float*       out = (float*)d_out;
    _Float16*    xh  = (_Float16*)d_ws;                                  // 2 MB
    _Float16*    xt  = xh + (size_t)B_DIM * C_DIM * N_DIM;               // 2 MB

    cvt_kernel<<<dim3(B_DIM * 64), dim3(256), 0, stream>>>(x, xh, xt);
    fa_kernel <<<dim3(B_DIM * 64), dim3(32 * WPB), 0, stream>>>(xh, xt, x, out);
}
